// Net_27401891348772
// MI455X (gfx1250) — matrix operand order repaired and checked by reading, not compiled, not run on hardware
//
#include <hip/hip_runtime.h>
#include <hip/hip_bf16.h>

#define NPTS  1024
#define KNN   16
#define BATCH 8
#define NK    (NPTS * KNN)
#define BNINV 0.99999499993750f   /* 1/sqrt(1+1e-5) */

typedef __attribute__((ext_vector_type(16))) __bf16 v16bf;
typedef __attribute__((ext_vector_type(8)))  float  v8f;

union BFrag { v16bf v; unsigned short u[16]; };

__device__ __forceinline__ unsigned short f2bf(float f) {
    union { float f; unsigned u; } c; c.f = f;
    unsigned r = c.u + 0x7FFFu + ((c.u >> 16) & 1u);   // round-to-nearest-even
    return (unsigned short)(r >> 16);
}
__device__ __forceinline__ float lrelu(float x) { return x >= 0.f ? x : 0.2f * x; }

// ---------------------------------------------------------------- kNN (top-16 by -dist^2)
__global__ void knn_kernel(const float* __restrict__ x, int* __restrict__ idx)
{
    int b = blockIdx.y;
    int n = blockIdx.x * blockDim.x + threadIdx.x;
    __shared__ float sx[3][NPTS];
    for (int t = threadIdx.x; t < 3 * NPTS; t += blockDim.x)
        sx[t / NPTS][t % NPTS] = x[b * 3 * NPTS + t];
    __syncthreads();
    float qx = sx[0][n], qy = sx[1][n], qz = sx[2][n];
    float best[KNN]; int bid[KNN];
#pragma unroll
    for (int k = 0; k < KNN; ++k) { best[k] = -3.4e38f; bid[k] = 0; }
    for (int m = 0; m < NPTS; ++m) {
        float dx = qx - sx[0][m], dy = qy - sx[1][m], dz = qz - sx[2][m];
        float pd = -(dx * dx + dy * dy + dz * dz);
        if (pd > best[KNN - 1]) {
            best[KNN - 1] = pd; bid[KNN - 1] = m;
#pragma unroll
            for (int j = KNN - 1; j > 0; --j) {       // single bubble pass (list was sorted)
                if (best[j] > best[j - 1]) {
                    float tv = best[j]; best[j] = best[j - 1]; best[j - 1] = tv;
                    int   ti = bid[j];  bid[j]  = bid[j - 1];  bid[j - 1]  = ti;
                }
            }
        }
    }
#pragma unroll
    for (int k = 0; k < KNN; ++k)
        idx[(b * NPTS + n) * KNN + k] = bid[k];
}

// ---------------------------------------------------------------- geo = graph_feature(x)
__global__ void geo_kernel(const float* __restrict__ x, const int* __restrict__ idx,
                           float* __restrict__ geo)
{
    int t = blockIdx.x * blockDim.x + threadIdx.x;   // b*NK + r
    if (t >= BATCH * NK) return;
    int b = t / NK, r = t % NK;
    int n = r >> 4;
    int p = idx[t];
    const float* xb = x + (long)b * 3 * NPTS;
    float* gb = geo + (long)b * 6 * NK;
#pragma unroll
    for (int c = 0; c < 3; ++c) {
        float ctr = xb[c * NPTS + n];
        float nbr = xb[c * NPTS + p];
        gb[(long)c * NK + r]       = nbr - ctr;
        gb[(long)(3 + c) * NK + r] = ctr;
    }
}

// ---------------------------------------------------------------- W1s[o, i*64+c] = sum_h w1[o*24+i*4+h, c]
__global__ void prep_w1s_kernel(const float* __restrict__ w1, float* __restrict__ w1s)
{
    int t = blockIdx.x * blockDim.x + threadIdx.x;
    if (t >= 64 * 384) return;
    int o = t / 384, q = t % 384;
    int i = q >> 6, c = q & 63;
    float s = 0.f;
#pragma unroll
    for (int h = 0; h < 4; ++h) s += w1[(o * 24 + i * 4 + h) * 64 + c];
    w1s[t] = s;
}

// ---------------------------------------------------------------- generic WMMA conv1x1
// 128 threads = 4 waves. Block tile: 64 co x 64 points. LDS-staged bf16 operands.
// Wave w owns co rows [co0b + 16w, +16); 4 accumulators cover the 64 points.
__global__ __launch_bounds__(128) void conv_wmma_kernel(
    const float* __restrict__ W, int Co, int Ci,
    const float* __restrict__ In, long in_bs, long in_cs,          // mode 0: In[b, k, p]
    const float* __restrict__ Xg, long xg_bs, long xg_cs, int Cin, // mode 1: graph feature of Xg[b, c, n]
    const int* __restrict__ idx,
    int mode,
    const float* __restrict__ gamma, const float* __restrict__ beta,
    float* __restrict__ Out, long out_bs, long out_cs,
    int out_max, int P)
{
    __shared__ unsigned short lds_a[64 * 32];   // [co_local][k_local]
    __shared__ unsigned short lds_b[32 * 64];   // [k_local][p_local]
    int b    = blockIdx.z;
    int bx   = blockIdx.x;
    int p0   = bx * 64;
    int co0b = blockIdx.y * 64;
    int t    = threadIdx.x;
    int wv = t >> 5, ln = t & 31, hf = ln >> 4, lm = ln & 15;

    v8f acc[4];
#pragma unroll
    for (int s = 0; s < 4; ++s) acc[s] = (v8f){0.f,0.f,0.f,0.f,0.f,0.f,0.f,0.f};

    int ktiles = (Ci + 31) >> 5;
    for (int kt = 0; kt < ktiles; ++kt) {
        int k0 = kt * 32;
        if (kt + 1 < ktiles) {                   // prefetch next k-tile (global_prefetch_b8)
            __builtin_prefetch((const char*)(W + (long)co0b * Ci + (k0 + 32)) + t * 64, 0, 0);
            if (mode == 0)
                __builtin_prefetch((const char*)(In + b * in_bs + (long)(k0 + 32) * in_cs + p0) + t * 64, 0, 0);
        }
        // ---- stage A (weights): 64 rows x 32 ks, coalesced along k ----
#pragma unroll
        for (int i = 0; i < 16; ++i) {
            int e = i * 128 + t;
            int r = e >> 5, k = k0 + (e & 31);
            lds_a[e] = (k < Ci) ? f2bf(W[(long)(co0b + r) * Ci + k]) : (unsigned short)0;
        }
        // ---- stage B (activations): 32 ks x 64 points, coalesced along p ----
        if (mode == 0) {
#pragma unroll
            for (int i = 0; i < 16; ++i) {
                int e = i * 128 + t;
                int k = k0 + (e >> 6), p = e & 63;
                lds_b[e] = (k < Ci) ? f2bf(In[b * in_bs + (long)k * in_cs + p0 + p])
                                    : (unsigned short)0;
            }
        } else {                                  // fused graph_feature gather
#pragma unroll
            for (int i = 0; i < 16; ++i) {
                int e = i * 128 + t;
                int k = k0 + (e >> 6), p = e & 63;
                int gp = p0 + p, n = gp >> 4, jj = gp & 15;
                float v = 0.f;
                if (k < Ci) {
                    int cc = (k < Cin) ? k : k - Cin;
                    const float* base = Xg + b * xg_bs + (long)cc * xg_cs;
                    float ctr = base[n];
                    if (k < Cin) { int pt = idx[(b * NPTS + n) * KNN + jj]; v = base[pt] - ctr; }
                    else         { v = ctr; }
                }
                lds_b[e] = f2bf(v);
            }
        }
        __syncthreads();
        // ---- fragments + 4 WMMAs (A reused 4x) ----
        BFrag a;
#pragma unroll
        for (int j = 0; j < 16; ++j) {            // ISA 16-bit A 16x32 layout
            int kl = hf * 8 + (j & 7) + ((j >> 3) << 4);
            a.u[j] = lds_a[(wv * 16 + lm) * 32 + kl];
        }
#pragma unroll
        for (int s = 0; s < 4; ++s) {
            BFrag bb;
#pragma unroll
            for (int j = 0; j < 16; ++j)          // lane = N column; K in table order
                bb.u[j] = lds_b[(hf * 8 + (j & 7) + ((j >> 3) << 4)) * 64 + s * 16 + lm];
            acc[s] = __builtin_amdgcn_wmma_f32_16x16x32_bf16(false, a.v, false, bb.v,
                                                             (short)0, acc[s], false, false);
        }
        __syncthreads();
    }
    // ---- epilogue: BN + lrelu (+ max over the 16 neighbors) ----
#pragma unroll
    for (int s = 0; s < 4; ++s) {
#pragma unroll
        for (int v = 0; v < 8; ++v) {
            int r = co0b + wv * 16 + v + 8 * hf;
            float val = acc[s][v] * (gamma[r] * BNINV) + beta[r];
            val = lrelu(val);
            if (out_max) {
#pragma unroll
                for (int off = 8; off >= 1; off >>= 1)
                    val = fmaxf(val, __shfl_xor(val, off, 16));
                if (lm == 0)
                    Out[b * out_bs + (long)r * out_cs + (bx * 4 + s)] = val;
            } else {
                Out[b * out_bs + (long)r * out_cs + (p0 + s * 16 + lm)] = val;
            }
        }
    }
}

// ---------------------------------------------------------------- MAK stage C:
// out[o,p] = lrelu( bn(W1s @ (geo*ym), go,bo) + bn(ws@geo, gs,bs) ), then max over k.
// Block tile: 64 co (all) x 64 points.
__global__ __launch_bounds__(128) void stagec_wmma_kernel(
    const float* __restrict__ W1s,                     // 64 x 384
    const float* __restrict__ ym,                      // (B,64,NK)
    const float* __restrict__ geo,                     // (B,6,NK)
    const float* __restrict__ wsd,                     // 64 x 6
    const float* __restrict__ go, const float* __restrict__ bo,
    const float* __restrict__ gs, const float* __restrict__ bs,
    float* __restrict__ Out, long out_bs, long out_cs) // (B, ., N) slice
{
    __shared__ unsigned short lds_a[64 * 32];
    __shared__ unsigned short lds_b[32 * 64];
    int b  = blockIdx.z;
    int bx = blockIdx.x;
    int p0 = bx * 64;
    int t  = threadIdx.x;
    int wv = t >> 5, ln = t & 31, hf = ln >> 4, lm = ln & 15;
    const float* geoB = geo + (long)b * 6 * NK;
    const float* ymB  = ym  + (long)b * 64 * NK;

    v8f acc[4];
#pragma unroll
    for (int s = 0; s < 4; ++s) acc[s] = (v8f){0.f,0.f,0.f,0.f,0.f,0.f,0.f,0.f};

    for (int kt = 0; kt < 12; ++kt) {                  // 384 / 32
        int k0 = kt * 32;
        if (kt + 1 < 12)
            __builtin_prefetch((const char*)(ymB + (long)(((k0 + 32) & 63)) * NK + p0) + t * 64, 0, 0);
        // ---- stage A (folded weights) ----
#pragma unroll
        for (int i = 0; i < 16; ++i) {
            int e = i * 128 + t;
            int r = e >> 5, k = k0 + (e & 31);
            lds_a[e] = f2bf(W1s[r * 384 + k]);
        }
        // ---- stage B: Z[q,p] = geo[i,p] * ym[c,p], q = i*64+c ----
#pragma unroll
        for (int i = 0; i < 16; ++i) {
            int e = i * 128 + t;
            int q = k0 + (e >> 6), p = p0 + (e & 63);
            int ii = q >> 6, c = q & 63;
            lds_b[e] = f2bf(geoB[(long)ii * NK + p] * ymB[(long)c * NK + p]);
        }
        __syncthreads();
        BFrag a;
#pragma unroll
        for (int j = 0; j < 16; ++j) {
            int kl = hf * 8 + (j & 7) + ((j >> 3) << 4);
            a.u[j] = lds_a[(wv * 16 + lm) * 32 + kl];
        }
#pragma unroll
        for (int s = 0; s < 4; ++s) {
            BFrag bb;
#pragma unroll
            for (int j = 0; j < 16; ++j)
                bb.u[j] = lds_b[(hf * 8 + (j & 7) + ((j >> 3) << 4)) * 64 + s * 16 + lm];
            acc[s] = __builtin_amdgcn_wmma_f32_16x16x32_bf16(false, a.v, false, bb.v,
                                                             (short)0, acc[s], false, false);
        }
        __syncthreads();
    }
#pragma unroll
    for (int s = 0; s < 4; ++s) {
#pragma unroll
        for (int v = 0; v < 8; ++v) {
            int r = wv * 16 + v + 8 * hf;
            int p = p0 + s * 16 + lm;
            float val = acc[s][v] * (go[r] * BNINV) + bo[r];
            float sum = 0.f;
#pragma unroll
            for (int i = 0; i < 6; ++i)
                sum += wsd[r * 6 + i] * geoB[(long)i * NK + p];
            val += sum * (gs[r] * BNINV) + bs[r];
            val = lrelu(val);
#pragma unroll
            for (int off = 8; off >= 1; off >>= 1)
                val = fmaxf(val, __shfl_xor(val, off, 16));
            if (lm == 0)
                Out[b * out_bs + (long)r * out_cs + (bx * 4 + s)] = val;
        }
    }
}

// ---------------------------------------------------------------- global max+mean pool over N
__global__ void pool_kernel(const float* __restrict__ x5, float* __restrict__ xp)
{
    int t = blockIdx.x * blockDim.x + threadIdx.x;     // b*1024 + c
    if (t >= BATCH * 1024) return;
    int b = t >> 10, c = t & 1023;
    const float* p = x5 + ((long)b * 1024 + c) * NPTS;
    float mx = -3.4e38f, sm = 0.f;
    for (int n = 0; n < NPTS; ++n) { float v = p[n]; mx = fmaxf(mx, v); sm += v; }
    xp[b * 2048 + c]        = mx;
    xp[b * 2048 + 1024 + c] = sm * (1.f / NPTS);
}

// ---------------------------------------------------------------- small FC (+bias, +BN, +lrelu)
__global__ void fc_kernel(const float* __restrict__ in, const float* __restrict__ w,
                          const float* __restrict__ bias,
                          const float* __restrict__ g, const float* __restrict__ bb,
                          float* __restrict__ out, int Ci, int Co, int act)
{
    int t = blockIdx.x * blockDim.x + threadIdx.x;
    if (t >= BATCH * Co) return;
    int b = t / Co, o = t % Co;
    const float* ip = in + (long)b * Ci;
    const float* wp = w + (long)o * Ci;
    float s = 0.f;
    for (int k = 0; k < Ci; ++k) s += ip[k] * wp[k];
    if (bias) s += bias[o];
    if (g)    s = s * (g[o] * BNINV) + bb[o];
    if (act)  s = lrelu(s);
    out[t] = s;
}

// ================================================================ launch
extern "C" void kernel_launch(void* const* d_in, const int* in_sizes, int n_in,
                              void* d_out, int out_size, void* d_ws, size_t ws_size,
                              hipStream_t stream)
{
    const float* x      = (const float*)d_in[0];
    const float* m1_w0  = (const float*)d_in[1];
    const float* m1_g0  = (const float*)d_in[2];
    const float* m1_b0  = (const float*)d_in[3];
    const float* m1_wm  = (const float*)d_in[4];
    const float* m1_gm  = (const float*)d_in[5];
    const float* m1_bm  = (const float*)d_in[6];
    const float* m1_w1  = (const float*)d_in[7];
    const float* m1_go  = (const float*)d_in[8];
    const float* m1_bo  = (const float*)d_in[9];
    const float* m1_ws  = (const float*)d_in[10];
    const float* m1_gs  = (const float*)d_in[11];
    const float* m1_bs  = (const float*)d_in[12];
    const float* m2_w0  = (const float*)d_in[13];
    const float* m2_g0  = (const float*)d_in[14];
    const float* m2_b0  = (const float*)d_in[15];
    const float* m2_wm  = (const float*)d_in[16];
    const float* m2_gm  = (const float*)d_in[17];
    const float* m2_bm  = (const float*)d_in[18];
    const float* m2_w1  = (const float*)d_in[19];
    const float* m2_go  = (const float*)d_in[20];
    const float* m2_bo  = (const float*)d_in[21];
    const float* m2_ws  = (const float*)d_in[22];
    const float* m2_gs  = (const float*)d_in[23];
    const float* m2_bs  = (const float*)d_in[24];
    const float* w3     = (const float*)d_in[25];
    const float* g3     = (const float*)d_in[26];
    const float* b3     = (const float*)d_in[27];
    const float* w4     = (const float*)d_in[28];
    const float* g4     = (const float*)d_in[29];
    const float* b4     = (const float*)d_in[30];
    const float* w5     = (const float*)d_in[31];
    const float* g5     = (const float*)d_in[32];
    const float* b5     = (const float*)d_in[33];
    const float* wl1    = (const float*)d_in[34];
    const float* bn4_g  = (const float*)d_in[35];
    const float* bn4_b  = (const float*)d_in[36];
    const float* wl2    = (const float*)d_in[37];
    const float* bl2    = (const float*)d_in[38];
    const float* bn5_g  = (const float*)d_in[39];
    const float* bn5_b  = (const float*)d_in[40];
    const float* wl3    = (const float*)d_in[41];
    const float* bl3    = (const float*)d_in[42];

    char* wsb = (char*)d_ws;
    size_t off = 0;
    auto alloc = [&](size_t bytes) -> void* {
        void* p = wsb + off; off += (bytes + 255) & ~(size_t)255; return p;
    };
    int*   idx  = (int*)  alloc((size_t)BATCH * NPTS * KNN * 4);
    float* geo  = (float*)alloc((size_t)BATCH * 6 * NK * 4);
    float* y0   = (float*)alloc((size_t)BATCH * 64 * NK * 4);
    float* ymb  = (float*)alloc((size_t)BATCH * 64 * NK * 4);
    float* xc   = (float*)alloc((size_t)BATCH * 512 * NPTS * 4);   // [x1|x2|x3|x4]
    float* x5b  = (float*)alloc((size_t)BATCH * 1024 * NPTS * 4);
    float* xp   = (float*)alloc((size_t)BATCH * 2048 * 4);
    float* xf1  = (float*)alloc((size_t)BATCH * 512 * 4);
    float* xf2  = (float*)alloc((size_t)BATCH * 256 * 4);
    float* w1s1 = (float*)alloc((size_t)64 * 384 * 4);
    float* w1s2 = (float*)alloc((size_t)64 * 384 * 4);

    const long XCB = (long)512 * NPTS;    // xc batch stride
    const long NKL = (long)NK;

    knn_kernel<<<dim3(NPTS / 128, BATCH), 128, 0, stream>>>(x, idx);
    geo_kernel<<<(BATCH * NK + 255) / 256, 256, 0, stream>>>(x, idx, geo);
    prep_w1s_kernel<<<(64 * 384 + 255) / 256, 256, 0, stream>>>(m1_w1, w1s1);
    prep_w1s_kernel<<<(64 * 384 + 255) / 256, 256, 0, stream>>>(m2_w1, w1s2);

    // ---- MAK 1 ----
    conv_wmma_kernel<<<dim3(NK / 64, 1, BATCH), 128, 0, stream>>>(
        m1_w0, 64, 6, geo, 6 * NKL, NKL,
        nullptr, 0, 0, 0, nullptr, 0,
        m1_g0, m1_b0, y0, 64 * NKL, NKL, 0, NK);
    conv_wmma_kernel<<<dim3(NK / 64, 1, BATCH), 128, 0, stream>>>(
        m1_wm, 64, 64, y0, 64 * NKL, NKL,
        nullptr, 0, 0, 0, nullptr, 0,
        m1_gm, m1_bm, ymb, 64 * NKL, NKL, 0, NK);
    stagec_wmma_kernel<<<dim3(NK / 64, 1, BATCH), 128, 0, stream>>>(
        w1s1, ymb, geo, m1_ws, m1_go, m1_bo, m1_gs, m1_bs,
        xc + 0 * NPTS, XCB, NPTS);                                  // -> x1

    // ---- MAK 2 (input = graph_feature(x1)) ----
    conv_wmma_kernel<<<dim3(NK / 64, 1, BATCH), 128, 0, stream>>>(
        m2_w0, 64, 128, nullptr, 0, 0,
        xc + 0 * NPTS, XCB, NPTS, 64, idx, 1,
        m2_g0, m2_b0, y0, 64 * NKL, NKL, 0, NK);
    conv_wmma_kernel<<<dim3(NK / 64, 1, BATCH), 128, 0, stream>>>(
        m2_wm, 64, 64, y0, 64 * NKL, NKL,
        nullptr, 0, 0, 0, nullptr, 0,
        m2_gm, m2_bm, ymb, 64 * NKL, NKL, 0, NK);
    stagec_wmma_kernel<<<dim3(NK / 64, 1, BATCH), 128, 0, stream>>>(
        w1s2, ymb, geo, m2_ws, m2_go, m2_bo, m2_gs, m2_bs,
        xc + 64 * NPTS, XCB, NPTS);                                 // -> x2

    // ---- x3 = lrelu(bn(w3 @ graph(x2))).max(k) ----
    conv_wmma_kernel<<<dim3(NK / 64, 2, BATCH), 128, 0, stream>>>(
        w3, 128, 128, nullptr, 0, 0,
        xc + 64 * NPTS, XCB, NPTS, 64, idx, 1,
        g3, b3, xc + 128 * NPTS, XCB, NPTS, 1, NK);

    // ---- x4 = lrelu(bn(w4 @ graph(x3))).max(k) ----
    conv_wmma_kernel<<<dim3(NK / 64, 4, BATCH), 128, 0, stream>>>(
        w4, 256, 256, nullptr, 0, 0,
        xc + 128 * NPTS, XCB, NPTS, 128, idx, 1,
        g4, b4, xc + 256 * NPTS, XCB, NPTS, 1, NK);

    // ---- x5 = lrelu(bn(w5 @ xc)) ----
    conv_wmma_kernel<<<dim3(NPTS / 64, 16, BATCH), 128, 0, stream>>>(
        w5, 1024, 512, xc, XCB, (long)NPTS,
        nullptr, 0, 0, 0, nullptr, 0,
        g5, b5, x5b, (long)1024 * NPTS, (long)NPTS, 0, NPTS);

    // ---- head ----
    pool_kernel<<<(BATCH * 1024 + 255) / 256, 256, 0, stream>>>(x5b, xp);
    fc_kernel<<<(BATCH * 512 + 255) / 256, 256, 0, stream>>>(
        xp, wl1, nullptr, bn4_g, bn4_b, xf1, 2048, 512, 1);
    fc_kernel<<<(BATCH * 256 + 255) / 256, 256, 0, stream>>>(
        xf1, wl2, bl2, bn5_g, bn5_b, xf2, 512, 256, 1);
    fc_kernel<<<(BATCH * 49 + 255) / 256, 256, 0, stream>>>(
        xf2, wl3, bl3, nullptr, nullptr, (float*)d_out, 256, 49, 0);
}
